// CausalGraphLayer_22050362098277
// MI455X (gfx1250) — compile-verified, matrix operand order checked by reading, not run on hardware
//
#include <hip/hip_runtime.h>
#include <math.h>

// Problem constants (from reference): B=2, N=2048, T=32, C=32, NUM_BASES=4, K_CURR=16
#define NN   2048
#define TT   32
#define CCH  32
#define KK   16
#define TILE (TT * CCH)   // 1024 elements per (b,n) tile

typedef __attribute__((ext_vector_type(2)))  float    v2f;
typedef __attribute__((ext_vector_type(8)))  float    v8f;
typedef __attribute__((ext_vector_type(16))) _Float16 v16h;

#if defined(__has_builtin)
#if __has_builtin(__builtin_amdgcn_wmma_f32_16x16x4_f32)
#define HAVE_WMMA_F32X4 1
#endif
#endif

// One block per (b, n). Wave 0 builds this node's 16(k) x 32(c) edge-weight
// tile with WMMA (exact f32, contraction over the 4 bases), scaled by adj,
// into LDS. Then all 256 threads gather neighbor tiles with float4 loads and
// accumulate: out[t,c] = tanh( sum_k z[b, idx[n,k], t, c] * w[n,k,c] ).
__global__ __launch_bounds__(256) void causal_graph_fused(
    const float* __restrict__ z,       // [B][N][T][C]
    const int*   __restrict__ nidx,    // [N][KK]
    const float* __restrict__ adj,     // [N][32] (use first 16)
    const float* __restrict__ bases,   // [4][N][32] (use first 16)
    const float* __restrict__ coeff,   // [C][4]
    float*       __restrict__ out)     // [B][N][T][C]
{
    __shared__ __align__(16) float sw[KK * CCH];  // w[k][c] for this n
    __shared__ int sidx[KK];

    const int tid = threadIdx.x;
    const int bn  = blockIdx.x;
    const int b   = bn >> 11;          // N = 2048
    const int n   = bn & (NN - 1);

    if (tid < KK) sidx[tid] = nidx[n * KK + tid];

    if (tid < 32) {  // wave 0 only; uniform per-wave branch -> EXEC all ones
        const int  lane = tid;
        const int  half = lane >> 4;   // 0: lanes 0-15, 1: lanes 16-31
        const int  l    = lane & 15;   // column index j == k (and row m within tile)
        const float adjv = adj[n * 32 + l];

#ifdef HAVE_WMMA_F32X4
        // B operand (4x16 f32): VGPR0 = B[K=0|2][j], VGPR1 = B[K=1|3][j]
        v2f Bm;
        Bm[0] = bases[(size_t)((half ? 2 : 0) * NN + n) * 32 + l];
        Bm[1] = bases[(size_t)((half ? 3 : 1) * NN + n) * 32 + l];
#pragma unroll
        for (int ct = 0; ct < 2; ++ct) {      // two 16-row c tiles (C=32)
            // A operand (16x4 f32): lane<16 -> K=0,1 ; lane>=16 -> K=2,3
            v2f Am;
            Am[0] = coeff[(ct * 16 + l) * 4 + (half ? 2 : 0)];
            Am[1] = coeff[(ct * 16 + l) * 4 + (half ? 3 : 1)];
            v8f d = {};
            d = __builtin_amdgcn_wmma_f32_16x16x4_f32(
                    false, Am, false, Bm, (short)0, d, false, false);
            // D: VGPR r holds M = r + half*8, column N = l
#pragma unroll
            for (int r = 0; r < 8; ++r) {
                const int c = ct * 16 + half * 8 + r;
                sw[l * CCH + c] = d[r] * adjv;
            }
        }
#else
        // Fallback: zero-padded f16 WMMA 16x16x32 (codegen-confirmed builtin).
        const bool lt = (lane < 16);
        v16h Bm;                                     // 32x16 f16 B, K>=4 zero
#pragma unroll
        for (int i = 0; i < 16; ++i) Bm[i] = (_Float16)0.f;
#pragma unroll
        for (int i = 0; i < 4; ++i)
            Bm[i] = lt ? (_Float16)bases[(size_t)(i * NN + n) * 32 + l]
                       : (_Float16)0.f;
#pragma unroll
        for (int ct = 0; ct < 2; ++ct) {
            v16h Am;
#pragma unroll
            for (int i = 0; i < 16; ++i) Am[i] = (_Float16)0.f;
#pragma unroll
            for (int i = 0; i < 4; ++i)
                Am[i] = lt ? (_Float16)coeff[(ct * 16 + l) * 4 + i]
                           : (_Float16)0.f;
            v8f d = {};
            d = __builtin_amdgcn_wmma_f32_16x16x32_f16(
                    false, Am, false, Bm, (short)0, d, false, false);
#pragma unroll
            for (int r = 0; r < 8; ++r) {
                const int c = ct * 16 + half * 8 + r;
                sw[l * CCH + c] = d[r] * adjv;
            }
        }
#endif
    }
    __syncthreads();

    // Gather + accumulate: each thread owns 4 consecutive (t,c) elements.
    const int pos = tid * 4;                  // 256 threads * 4 = 1024 = T*C
    const int c0  = pos & (CCH - 1);          // channel of first element
    const float* zb = z + (size_t)b * NN * TILE;

    float4 acc = make_float4(0.f, 0.f, 0.f, 0.f);
#pragma unroll
    for (int k = 0; k < KK; ++k) {
        const float4 zv = *(const float4*)(zb + (size_t)sidx[k] * TILE + pos);
        const float4 wv = *(const float4*)(&sw[k * CCH + c0]);   // ds_load_b128
        acc.x = fmaf(zv.x, wv.x, acc.x);
        acc.y = fmaf(zv.y, wv.y, acc.y);
        acc.z = fmaf(zv.z, wv.z, acc.z);
        acc.w = fmaf(zv.w, wv.w, acc.w);
    }

    float4 r;
    r.x = tanhf(acc.x);
    r.y = tanhf(acc.y);
    r.z = tanhf(acc.z);
    r.w = tanhf(acc.w);
    *(float4*)(out + (size_t)bn * TILE + pos) = r;
}

extern "C" void kernel_launch(void* const* d_in, const int* in_sizes, int n_in,
                              void* d_out, int out_size, void* d_ws, size_t ws_size,
                              hipStream_t stream) {
    (void)in_sizes; (void)n_in; (void)out_size; (void)d_ws; (void)ws_size;
    const float* z     = (const float*)d_in[0];   // (B,N,T,C) f32
    const int*   nidx  = (const int*)  d_in[1];   // (N,16) i32
    const float* adj   = (const float*)d_in[2];   // (N,32) f32
    const float* bases = (const float*)d_in[3];   // (4,N,32) f32
    const float* coeff = (const float*)d_in[4];   // (32,4) f32
    float*       out   = (float*)d_out;           // (B,N,T,C) f32

    dim3 grid(2 * NN);   // one block per (b, n)
    dim3 block(256);     // 8 waves (wave32)
    causal_graph_fused<<<grid, block, 0, stream>>>(z, nidx, adj, bases, coeff, out);
}
